// BasicModel_24936580121158
// MI455X (gfx1250) — matrix operand order repaired and checked by reading, not compiled, not run on hardware
//
#include <hip/hip_runtime.h>
#include <hip/hip_bf16.h>
#include <math.h>

typedef __attribute__((ext_vector_type(16))) _Float16 v16h;
typedef __attribute__((ext_vector_type(8)))  _Float16 v8h;
typedef __attribute__((ext_vector_type(8)))  float    v8f;

#define THREADS 256
// Workspace layout (bytes). Required ws_size ~= 119 MB.
//  act1  f16 [600][64][1024] @ 0          (78,643,200)
//  act2  f16 [600][64][512]  @ 78,643,200 (39,321,600)
//  act3  f16 [600][64][256]  aliases act1 (act1 dead by then)
//  h_all f32 [600][8192]     aliases act2 (act2 dead by then)
//  packed weights, bn scale/shift, prob, protos at the tail.

// ---------------- prep: fold conv-bias + BN into scale/shift ----------------
__global__ void bn_prep_kernel(const float* __restrict__ g, const float* __restrict__ beta,
                               const float* __restrict__ mean, const float* __restrict__ var,
                               const float* __restrict__ bias,
                               float* __restrict__ scale, float* __restrict__ shift) {
  int c = threadIdx.x;
  if (c < 64) {
    float inv = g[c] * rsqrtf(var[c] + 1e-5f);
    scale[c] = inv;
    shift[c] = (bias[c] - mean[c]) * inv + beta[c];
  }
}

// repack w (64,64,3) f32 -> f16 A[m][k], k = dk*64 + cin
__global__ void wpack_kernel(const float* __restrict__ w, _Float16* __restrict__ out) {
  int idx = blockIdx.x * blockDim.x + threadIdx.x;
  if (idx >= 64 * 192) return;
  int cout = idx / 192, r = idx % 192, dk = r / 64, cin = r % 64;
  out[idx] = (_Float16)w[cout * 192 + cin * 3 + dk];
}

// ---------------- stage 1: Cin=1 conv+BN+ReLU+pool (scalar, tiny) ----------------
__global__ __launch_bounds__(THREADS) void stage1_kernel(
    const float* __restrict__ xs, const float* __restrict__ xu, const float* __restrict__ xq,
    const float* __restrict__ w, const float* __restrict__ scale, const float* __restrict__ shift,
    _Float16* __restrict__ act1) {
  int idx = blockIdx.x * THREADS + threadIdx.x;
  if (idx >= 600 * 64 * 1024) return;
  int lp = idx & 1023;
  int co = (idx >> 10) & 63;
  int s  = idx >> 16;
  int b = s / 150, j = s % 150;
  const float* x;
  if (j < 25)      x = xs + (size_t)(b * 25 + j) * 2048;
  else if (j < 75) x = xu + (size_t)(b * 50 + (j - 25)) * 2048;
  else             x = xq + (size_t)(b * 75 + (j - 75)) * 2048;
  float w0 = w[co * 3 + 0], w1 = w[co * 3 + 1], w2 = w[co * 3 + 2];
  float sc = scale[co], sh = shift[co];
  int c0 = 2 * lp;
  float xm1 = (c0 - 1 >= 0)   ? x[c0 - 1] : 0.f;
  float x0  = x[c0];
  float x1  = x[c0 + 1];
  float x2  = (c0 + 2 < 2048) ? x[c0 + 2] : 0.f;
  float v0 = w0 * xm1 + w1 * x0 + w2 * x1;
  float v1 = w0 * x0  + w1 * x1 + w2 * x2;
  float y0 = fmaxf(v0 * sc + sh, 0.f);
  float y1 = fmaxf(v1 * sc + sh, 0.f);
  act1[(size_t)s * 65536 + co * 1024 + lp] = (_Float16)fmaxf(y0, y1);
}

// ---------------- WMMA conv stages 2-4 ----------------
// GEMM per sample: M=64 (cout), K=192 (dk*64+cin), N=Lin columns; fused BN/ReLU/pool2.
// 8 waves/block: wave = (mtile 0..3) x (ntile 0..1); block loops over a range of
// 32-column tiles so the register-resident A fragments amortize.
template <bool OUT_F32>
__global__ __launch_bounds__(THREADS) void conv_wmma_kernel(
    const _Float16* __restrict__ actIn, const _Float16* __restrict__ Wp,
    const float* __restrict__ bnScale, const float* __restrict__ bnShift,
    _Float16* __restrict__ outH, float* __restrict__ outF, int Lin) {
  // transposed input tile: [col 0..33][ch 0..63], row stride padded to 72 halves
  __shared__ __attribute__((aligned(32))) _Float16 sInT[34 * 72];

  const int tid  = threadIdx.x;
  const int samp = blockIdx.y;
  const int wave = tid >> 5, lane = tid & 31;
  const int g = lane >> 4, ln = lane & 15;
  const int mbase = (wave & 3) * 16;   // M tile (cout)
  const int nbase = (wave >> 2) * 16;  // N tile (conv column within 32-col tile)
  const int nTiles = Lin >> 5;
  const int Lout = Lin >> 1;

  // A fragments (weights) are loop-invariant: hoist into registers.
  // f16 A 16x32 layout: lane = M; halves h<8 -> K = 8g + h ; h>=8 -> K = 16 + 8g + (h-8).
  v16h afrag[6];
  {
    const _Float16* wrow = Wp + (size_t)(mbase + ln) * 192;
#pragma unroll
    for (int chunk = 0; chunk < 6; ++chunk) {
      v8h alo = *(const v8h*)(wrow + chunk * 32 + g * 8);
      v8h ahi = *(const v8h*)(wrow + chunk * 32 + 16 + g * 8);
#pragma unroll
      for (int h = 0; h < 8; ++h) { afrag[chunk][h] = alo[h]; afrag[chunk][h + 8] = ahi[h]; }
    }
  }
  // per-lane BN coefficients for the 8 rows this lane accumulates (M = mbase + 8g + r)
  float sc[8], sh[8];
#pragma unroll
  for (int r = 0; r < 8; ++r) {
    int m = mbase + g * 8 + r;
    sc[r] = bnScale[m];
    sh[r] = bnShift[m];
  }

  const _Float16* base = actIn + (size_t)samp * 64 * Lin;
  _Float16* oH = outH ? outH + (size_t)samp * 64 * Lout : (_Float16*)0;
  float*    oF = outF ? outF + (size_t)samp * 64 * Lout : (float*)0;

  const int nPerBlk = nTiles / gridDim.x;
  const int t0 = blockIdx.x * nPerBlk;

  for (int t = t0; t < t0 + nPerBlk; ++t) {
    const int blockCol = t * 32;
    // stage input tile (transposed, with halo, zero-padded at edges)
    for (int i = tid; i < 64 * 34; i += THREADS) {
      int ch = i / 34, lc = i % 34;
      int gc = blockCol - 1 + lc;
      sInT[lc * 72 + ch] = (gc >= 0 && gc < Lin) ? base[(size_t)ch * Lin + gc] : (_Float16)0.f;
    }
    __syncthreads();

    v8f acc;
#pragma unroll
    for (int r = 0; r < 8; ++r) acc[r] = 0.f;

#pragma unroll
    for (int chunk = 0; chunk < 6; ++chunk) {
      // B 32x16 layout: half h<8 -> K = 8g + h ; h>=8 -> K = 16 + 8g + (h-8); N = ln. Within this run dk is
      // constant and cin is contiguous -> two 16B LDS loads.
      int kb0 = chunk * 32 + g * 8;
      int dk = kb0 >> 6, cin0 = kb0 & 63;
      const _Float16* p = sInT + (nbase + ln + dk) * 72 + cin0;
      v8h blo = *(const v8h*)p;
      v8h bhi = *(const v8h*)(p + 16);
      v16h bfrag;
#pragma unroll
      for (int h = 0; h < 8; ++h) { bfrag[h] = blo[h]; bfrag[h + 8] = bhi[h]; }
      acc = __builtin_amdgcn_wmma_f32_16x16x32_f16(false, afrag[chunk], false, bfrag,
                                                   (short)0, acc, false, false);
    }

    // epilogue: BN + ReLU, pool adjacent columns via cross-lane max.
    // D layout: acc[r] holds (M = mbase+8g+r, N = ln); column pairs = lane pairs.
    float pooled[8];
#pragma unroll
    for (int r = 0; r < 8; ++r) {
      float y = fmaxf(acc[r] * sc[r] + sh[r], 0.f);
      float o = __shfl_xor(y, 1, 32);
      pooled[r] = fmaxf(y, o);
    }
    int pc  = ln >> 1;                              // pooled column 0..7
    int col = (blockCol + nbase) / 2 + pc;          // global pooled column
    int rsel = (ln & 1) * 4;                        // even lanes store r 0..3, odd 4..7
#pragma unroll
    for (int i2 = 0; i2 < 4; ++i2) {
      int r = rsel + i2;
      int m = mbase + g * 8 + r;
      size_t o = (size_t)m * Lout + col;
      if (OUT_F32) oF[o] = pooled[r];
      else         oH[o] = (_Float16)pooled[r];
    }
    __syncthreads();
  }
}

// ---------------- squeeze-excite (per sample), in place ----------------
template <typename T>
__global__ __launch_bounds__(THREADS) void se_kernel(
    T* __restrict__ act, int L,
    const float* __restrict__ w1, const float* __restrict__ b1,
    const float* __restrict__ w2, const float* __restrict__ b2) {
  __shared__ float sPart[4][64];
  __shared__ float sMean[64];
  __shared__ float sZ[4];
  __shared__ float sG[64];
  const int tid = threadIdx.x;
  T* base = act + (size_t)blockIdx.x * 64 * L;
  int c = tid & 63, part = tid >> 6;
  float acc = 0.f;
  for (int l = part; l < L; l += 4) acc += (float)base[c * L + l];
  sPart[part][c] = acc;
  __syncthreads();
  if (tid < 64)
    sMean[tid] = (sPart[0][tid] + sPart[1][tid] + sPart[2][tid] + sPart[3][tid]) / (float)L;
  __syncthreads();
  if (tid < 4) {
    float z = b1[tid];
    for (int cc = 0; cc < 64; ++cc) z += sMean[cc] * w1[tid * 64 + cc];
    sZ[tid] = fmaxf(z, 0.f);
  }
  __syncthreads();
  if (tid < 64) {
    float gg = b2[tid];
    for (int jj = 0; jj < 4; ++jj) gg += sZ[jj] * w2[tid * 4 + jj];
    sG[tid] = 1.f / (1.f + expf(-gg));
  }
  __syncthreads();
  for (int e = tid; e < 64 * L; e += THREADS) {
    int cc = e / L;
    base[e] = (T)((float)base[e] * sG[cc]);
  }
}

// ---------------- clustering ----------------
__global__ void prob_init_kernel(const int* __restrict__ ys, float* __restrict__ prob) {
  int idx = blockIdx.x * blockDim.x + threadIdx.x;
  if (idx >= 4 * 75 * 5) return;
  int b = idx / 375, r = (idx / 5) % 75, k = idx % 5;
  float v = 0.f;
  if (r < 25) v = (ys[b * 25 + r] == k) ? 1.f : 0.f;
  prob[idx] = v;
}

__global__ __launch_bounds__(THREADS) void protos_kernel(
    const float* __restrict__ h, const float* __restrict__ prob,
    float* __restrict__ protos, int nRows) {
  int d = blockIdx.x * THREADS + threadIdx.x;     // 0..8191
  int b = blockIdx.y / 5, k = blockIdx.y % 5;
  float num = 0.f, den = 0.f;
  for (int n = 0; n < nRows; ++n) {
    float p = prob[(b * 75 + n) * 5 + k];
    den += p;
    num += p * h[((size_t)(b * 150 + n)) * 8192 + d];
  }
  protos[((size_t)(b * 5 + k)) * 8192 + d] = num / (den + 1e-8f);
}

// blocks = B*nPer; h row = b*150 + rowOff + n.
// probOut != null: softmax(-d2) -> prob rows [rowOff..); else logitsOut = -d2.
__global__ __launch_bounds__(THREADS) void dist_kernel(
    const float* __restrict__ h, const float* __restrict__ protos,
    float* __restrict__ probOut, float* __restrict__ logitsOut,
    int rowOff, int nPer) {
  __shared__ float red[THREADS];
  __shared__ float d2[5];
  int b = blockIdx.x / nPer, n = blockIdx.x % nPer;
  const float* hv = h + ((size_t)(b * 150 + rowOff + n)) * 8192;
  float acc[5] = {0.f, 0.f, 0.f, 0.f, 0.f};
  for (int d = threadIdx.x; d < 8192; d += THREADS) {
    float x = hv[d];
#pragma unroll
    for (int k = 0; k < 5; ++k) {
      float diff = x - protos[((size_t)(b * 5 + k)) * 8192 + d];
      acc[k] += diff * diff;
    }
  }
  for (int k = 0; k < 5; ++k) {
    red[threadIdx.x] = acc[k];
    __syncthreads();
    for (int s = THREADS / 2; s > 0; s >>= 1) {
      if (threadIdx.x < s) red[threadIdx.x] += red[threadIdx.x + s];
      __syncthreads();
    }
    if (threadIdx.x == 0) d2[k] = red[0];
    __syncthreads();
  }
  if (threadIdx.x == 0) {
    if (probOut) {
      float m = -d2[0];
      for (int k = 1; k < 5; ++k) m = fmaxf(m, -d2[k]);
      float s = 0.f, e[5];
      for (int k = 0; k < 5; ++k) { e[k] = expf(-d2[k] - m); s += e[k]; }
      for (int k = 0; k < 5; ++k)
        probOut[((size_t)(b * 75 + rowOff + n)) * 5 + k] = e[k] / s;
    } else {
      for (int k = 0; k < 5; ++k)
        logitsOut[((size_t)(b * nPer + n)) * 5 + k] = -d2[k];
    }
  }
}

__global__ void hqcopy_kernel(const float* __restrict__ h, float* __restrict__ out) {
  int i = blockIdx.x * blockDim.x + threadIdx.x;   // float4 index
  const int per = 75 * 8192 / 4;                   // 153600
  if (i >= 4 * per) return;
  int b = i / per, r = i % per;
  const float4* src = (const float4*)(h + ((size_t)(b * 150 + 75)) * 8192);
  float4* dst = (float4*)(out + 1500);             // 6000 B offset, 16B aligned
  dst[(size_t)b * per + r] = src[r];
}

extern "C" void kernel_launch(void* const* d_in, const int* in_sizes, int n_in,
                              void* d_out, int out_size, void* d_ws, size_t ws_size,
                              hipStream_t stream) {
  (void)in_sizes; (void)n_in; (void)out_size; (void)ws_size;
  const float* xs = (const float*)d_in[0];
  const float* xu = (const float*)d_in[1];
  const float* xq = (const float*)d_in[2];
  const int*   ys = (const int*)d_in[3];
  // d_in[4] = n_class (always 5 here)
  const float *W[4], *BC[4], *G[4], *BB[4], *M[4], *V[4];
  for (int i = 0; i < 4; ++i) {
    int o = 5 + 6 * i;
    W[i]  = (const float*)d_in[o + 0];
    BC[i] = (const float*)d_in[o + 1];
    G[i]  = (const float*)d_in[o + 2];
    BB[i] = (const float*)d_in[o + 3];
    M[i]  = (const float*)d_in[o + 4];
    V[i]  = (const float*)d_in[o + 5];
  }
  const float* sew1 = (const float*)d_in[29];
  const float* seb1 = (const float*)d_in[30];
  const float* sew2 = (const float*)d_in[31];
  const float* seb2 = (const float*)d_in[32];
  float* out = (float*)d_out;

  char* ws = (char*)d_ws;
  const size_t OFF_ACT1 = 0;
  const size_t OFF_ACT2 = 78643200;
  const size_t OFF_WP   = 117964800;
  const size_t OFF_BN   = OFF_WP + 3 * 24576;
  const size_t OFF_PROB = OFF_BN + 4 * 512;
  const size_t OFF_PROT = OFF_PROB + 6144;
  _Float16* act1 = (_Float16*)(ws + OFF_ACT1);
  _Float16* act2 = (_Float16*)(ws + OFF_ACT2);
  _Float16* act3 = (_Float16*)(ws + OFF_ACT1);  // alias (act1 dead)
  float*    hall = (float*)   (ws + OFF_ACT2);  // alias (act2 dead)
  _Float16* wp   = (_Float16*)(ws + OFF_WP);
  float*    bn   = (float*)   (ws + OFF_BN);    // per stage: 64 scale + 64 shift
  float*    prob = (float*)   (ws + OFF_PROB);
  float*    prot = (float*)   (ws + OFF_PROT);

  for (int i = 0; i < 4; ++i)
    bn_prep_kernel<<<1, 64, 0, stream>>>(G[i], BB[i], M[i], V[i], BC[i],
                                         bn + i * 128, bn + i * 128 + 64);
  for (int i = 0; i < 3; ++i)
    wpack_kernel<<<48, 256, 0, stream>>>(W[i + 1], wp + (size_t)i * 64 * 192);

  stage1_kernel<<<153600, 256, 0, stream>>>(xs, xu, xq, W[0], bn + 0, bn + 64, act1);
  conv_wmma_kernel<false><<<dim3(4, 600), 256, 0, stream>>>(act1, wp,         bn + 128, bn + 192, act2, nullptr, 1024);
  conv_wmma_kernel<false><<<dim3(4, 600), 256, 0, stream>>>(act2, wp + 12288, bn + 256, bn + 320, act3, nullptr, 512);
  se_kernel<_Float16><<<600, 256, 0, stream>>>(act3, 256, sew1, seb1, sew2, seb2);
  conv_wmma_kernel<true><<<dim3(4, 600), 256, 0, stream>>>(act3, wp + 24576,  bn + 384, bn + 448, nullptr, hall, 256);
  se_kernel<float><<<600, 256, 0, stream>>>(hall, 128, sew1, seb1, sew2, seb2);

  prob_init_kernel<<<6, 256, 0, stream>>>(ys, prob);
  protos_kernel<<<dim3(32, 20), 256, 0, stream>>>(hall, prob, prot, 25);
  for (int it = 0; it < 3; ++it) {
    dist_kernel<<<200, 256, 0, stream>>>(hall, prot, prob, nullptr, 25, 50);
    protos_kernel<<<dim3(32, 20), 256, 0, stream>>>(hall, prob, prot, 75);
  }
  dist_kernel<<<300, 256, 0, stream>>>(hall, prot, nullptr, out, 75, 75);
  hqcopy_kernel<<<2400, 256, 0, stream>>>(hall, out);
}
